// TorchLMHeadSimPO_47476568490607
// MI455X (gfx1250) — compile-verified, matrix operand order checked by reading, not run on hardware
//
#include <hip/hip_runtime.h>
#include <hip/hip_bf16.h>
#include <math.h>

// ---------------------------------------------------------------------------
// SimPO LM-head loss, fused GEMM + online softmax partials, bf16 WMMA (CDNA5)
// logits = x @ W^T (M=2048, K=4096, V=32000); per-token logp = logit[y]-logZ;
// per-sequence averaging + SimPO scalar epilogue.
// K-loop is unrolled x2 with ping-pong fragment buffers (no register copies)
// and a single per-lane base pointer per operand; all sub-tile displacements
// are folded into the 24-bit global_load immediate offsets.
// ---------------------------------------------------------------------------

#define HDIM 4096
#define VDIM 32000
#define MTOT 2048          // 8 * 256 tokens
#define SEQS 8
#define TSEQ 256
#define BMT  128           // M tile
#define BNT  128           // N tile
#define BKT  32            // K step (bf16 WMMA K)
#define NVT  (VDIM / BNT)  // 250 vocab tiles
#define NMT  (MTOT / BMT)  // 16 token tiles
#define IGNORE_IDX (-100)
#define BETA_C   0.1f
#define ALPHA_C  1.0f
#define GAMMA_C  0.5f

typedef __attribute__((ext_vector_type(16))) __bf16 v16bf;
typedef __attribute__((ext_vector_type(8)))  __bf16 v8bf;
typedef __attribute__((ext_vector_type(4)))  __bf16 v4bf;
typedef __attribute__((ext_vector_type(8)))  float  v8f;
typedef __attribute__((ext_vector_type(4)))  float  v4f;

// ---------------------------------------------------------------------------
// fp32 -> bf16 conversion (vectorized x4)
// ---------------------------------------------------------------------------
__global__ void simpo_cvt_bf16(const float* __restrict__ src,
                               __bf16* __restrict__ dst, long n) {
  long i = ((long)blockIdx.x * blockDim.x + threadIdx.x) * 4;
  if (i + 4 <= n) {
    v4f v = *reinterpret_cast<const v4f*>(src + i);
    v4bf o;
    o[0] = (__bf16)v[0]; o[1] = (__bf16)v[1];
    o[2] = (__bf16)v[2]; o[3] = (__bf16)v[3];
    *reinterpret_cast<v4bf*>(dst + i) = o;
  }
}

// ---------------------------------------------------------------------------
// Fragment loaders per ISA 7.12.2 (16-bit A 16x32): lane<16 -> K {0..7,16..23},
// lane>=16 -> K {8..15,24..31}. Base pointer already includes the lane's
// kh*8 shift; `eo` is a compile-time-constant element offset so sub-tile
// displacements become global_load immediate offsets.
// ---------------------------------------------------------------------------
__device__ inline v16bf frag_bf16(const __bf16* p, int eo) {
  const v8bf lo = *reinterpret_cast<const v8bf*>(p + eo);
  const v8bf hi = *reinterpret_cast<const v8bf*>(p + eo + 16);
  v16bf f;
#pragma unroll
  for (int e = 0; e < 8; ++e) { f[e] = lo[e]; f[8 + e] = hi[e]; }
  return f;
}

__device__ inline v16bf frag_f32(const float* p, int eo) {
  const v4f l0 = *reinterpret_cast<const v4f*>(p + eo);
  const v4f l1 = *reinterpret_cast<const v4f*>(p + eo + 4);
  const v4f h0 = *reinterpret_cast<const v4f*>(p + eo + 16);
  const v4f h1 = *reinterpret_cast<const v4f*>(p + eo + 20);
  v16bf f;
#pragma unroll
  for (int e = 0; e < 4; ++e) {
    f[e]      = (__bf16)l0[e]; f[4 + e]  = (__bf16)l1[e];
    f[8 + e]  = (__bf16)h0[e]; f[12 + e] = (__bf16)h1[e];
  }
  return f;
}

// ---------------------------------------------------------------------------
// Fused GEMM tile + softmax partials.
// Grid: (NMT=16, NVT=250)  -> M fastest so x stays L2-resident, W streams once.
// Block: 256 threads = 8 waves; wave tile 64(M) x 32(N) = 4x2 WMMA tiles.
// ---------------------------------------------------------------------------
template <bool CONV>
__global__ __launch_bounds__(256)
void simpo_gemm_partials(const float* __restrict__ Xf, const float* __restrict__ Wf,
                         const __bf16* __restrict__ Xb, const __bf16* __restrict__ Wb,
                         const int* __restrict__ Y,
                         float* __restrict__ part_max, float* __restrict__ part_sum,
                         float* __restrict__ tgt) {
  constexpr int LDSN = BNT + 4;                 // padded stride, 132 floats
  __shared__ float smem[BMT * LDSN];            // 67.6 KB of the 320 KB WGP LDS

  const int m0   = blockIdx.x * BMT;
  const int n0   = blockIdx.y * BNT;
  const int tid  = threadIdx.x;
  const int wave = tid >> 5;
  const int lane = tid & 31;
  const int waveM = (wave & 1) * 64;
  const int waveN = (wave >> 1) * 32;
  const int rl = lane & 15;   // row (A) / col (B) within 16
  const int kh = lane >> 4;   // K-half select

  // Per-lane base pointers (include kh*8 K-shift); everything else is an
  // immediate: A sub-tile i -> +i*16*HDIM, B sub-tile j -> +j*16*HDIM,
  // hi half -> +16, K-steps -> +32/+64 elements.
  const __bf16* aB = Xb + (size_t)(m0 + waveM + rl) * HDIM + kh * 8;
  const __bf16* bB = Wb + (size_t)(n0 + waveN + rl) * HDIM + kh * 8;
  const float*  aF = Xf + (size_t)(m0 + waveM + rl) * HDIM + kh * 8;
  const float*  bF = Wf + (size_t)(n0 + waveN + rl) * HDIM + kh * 8;

  auto loadA = [&](int eo) -> v16bf {
    if constexpr (CONV) return frag_bf16(aB, eo); else return frag_f32(aF, eo);
  };
  auto loadB = [&](int eo) -> v16bf {
    if constexpr (CONV) return frag_bf16(bB, eo); else return frag_f32(bF, eo);
  };

  v8f acc[4][2] = {};
  v16bf a0[4], b0[2], a1[4], b1[2];

  // Prologue: buffer 0 <- K-step 0.
#pragma unroll
  for (int i = 0; i < 4; ++i) a0[i] = loadA(i * 16 * HDIM);
#pragma unroll
  for (int j = 0; j < 2; ++j) b0[j] = loadB(j * 16 * HDIM);

  // Main loop, 2 K-steps per iteration; ping-pong buffers, zero reg copies.
  for (int k0 = 0; k0 < HDIM; k0 += 2 * BKT) {
    if constexpr (CONV) {   // prefetch W ~4 K-steps ahead (streamed operand)
      __builtin_prefetch((const void*)(bB + 128), 0, 1);
      __builtin_prefetch((const void*)(bB + 16 * HDIM + 128), 0, 1);
    } else {
      __builtin_prefetch((const void*)(bF + 128), 0, 1);
      __builtin_prefetch((const void*)(bF + 16 * HDIM + 128), 0, 1);
    }
    // buffer 1 <- K-step k0+BKT (always valid: HDIM/BKT is even)
#pragma unroll
    for (int i = 0; i < 4; ++i) a1[i] = loadA(i * 16 * HDIM + BKT);
#pragma unroll
    for (int j = 0; j < 2; ++j) b1[j] = loadB(j * 16 * HDIM + BKT);
#pragma unroll
    for (int i = 0; i < 4; ++i)
#pragma unroll
      for (int j = 0; j < 2; ++j)
        acc[i][j] = __builtin_amdgcn_wmma_f32_16x16x32_bf16(
            false, a0[i], false, b0[j], (short)0, acc[i][j], false, false);

    if (k0 + 2 * BKT < HDIM) {  // buffer 0 <- K-step k0+2*BKT
#pragma unroll
      for (int i = 0; i < 4; ++i) a0[i] = loadA(i * 16 * HDIM + 2 * BKT);
#pragma unroll
      for (int j = 0; j < 2; ++j) b0[j] = loadB(j * 16 * HDIM + 2 * BKT);
    }
#pragma unroll
    for (int i = 0; i < 4; ++i)
#pragma unroll
      for (int j = 0; j < 2; ++j)
        acc[i][j] = __builtin_amdgcn_wmma_f32_16x16x32_bf16(
            false, a1[i], false, b1[j], (short)0, acc[i][j], false, false);

    aB += 2 * BKT; bB += 2 * BKT; aF += 2 * BKT; bF += 2 * BKT;
  }

  // Spill C tiles to LDS. C layout: VGPR r, lane L -> (M = r + 8*(L>>4), N = L&15)
#pragma unroll
  for (int i = 0; i < 4; ++i)
#pragma unroll
    for (int j = 0; j < 2; ++j) {
      float* dst = &smem[(waveM + i * 16 + 8 * kh) * LDSN + waveN + j * 16 + rl];
#pragma unroll
      for (int r = 0; r < 8; ++r) dst[r * LDSN] = acc[i][j][r];
    }
  __syncthreads();

  // Row-wise (max, sum-exp) over the 128 columns of this tile.
  // 2 threads per row; pairs are adjacent lanes -> combine with shfl_xor(1).
  const int row  = tid >> 1;
  const int half = tid & 1;
  const float* rp = &smem[row * LDSN + half * 64];
  float mx = -INFINITY;
#pragma unroll 8
  for (int c = 0; c < 64; ++c) mx = fmaxf(mx, rp[c]);
  mx = fmaxf(mx, __shfl_xor(mx, 1));
  float se = 0.f;
#pragma unroll 8
  for (int c = 0; c < 64; ++c) se += __expf(rp[c] - mx);
  se += __shfl_xor(se, 1);

  if (half == 0) {
    const int mt = m0 + row;
    part_max[(size_t)mt * NVT + blockIdx.y] = mx;
    part_sum[(size_t)mt * NVT + blockIdx.y] = se;
    int yv = Y[mt];
    int ys = (yv == IGNORE_IDX) ? 0 : yv;
    if (ys >= n0 && ys < n0 + BNT)
      tgt[mt] = smem[row * LDSN + (ys - n0)];
  }
}

// ---------------------------------------------------------------------------
// Per-token log-sum-exp merge: one block per token, 256 threads over 250 tiles.
// ---------------------------------------------------------------------------
__global__ __launch_bounds__(256)
void simpo_logz(const float* __restrict__ part_max, const float* __restrict__ part_sum,
                const float* __restrict__ tgt, float* __restrict__ logp) {
  __shared__ float red[256];
  const int m = blockIdx.x;
  const int tid = threadIdx.x;
  float mx = (tid < NVT) ? part_max[(size_t)m * NVT + tid] : -INFINITY;
  red[tid] = mx;
  __syncthreads();
  for (int s = 128; s > 0; s >>= 1) {
    if (tid < s) red[tid] = fmaxf(red[tid], red[tid + s]);
    __syncthreads();
  }
  const float M = red[0];
  __syncthreads();
  float sv = (tid < NVT)
                 ? part_sum[(size_t)m * NVT + tid] * __expf(mx - M)
                 : 0.f;
  red[tid] = sv;
  __syncthreads();
  for (int s = 128; s > 0; s >>= 1) {
    if (tid < s) red[tid] += red[tid + s];
    __syncthreads();
  }
  if (tid == 0) logp[m] = tgt[m] - (M + logf(red[0]));
}

// ---------------------------------------------------------------------------
// Final SimPO scalar: 8 waves, one per sequence.
// ---------------------------------------------------------------------------
__device__ inline float log_sigmoid_f(float x) {
  return (x < 0.f) ? (x - log1pf(expf(x))) : (-log1pf(expf(-x)));
}

__global__ __launch_bounds__(256)
void simpo_final(const float* __restrict__ logp, const int* __restrict__ Y,
                 float* __restrict__ out) {
  __shared__ float ssum[SEQS];
  __shared__ float scnt[SEQS];
  const int tid = threadIdx.x;
  const int seq = tid >> 5;
  const int lane = tid & 31;
  float sum = 0.f, cnt = 0.f;
  for (int t = lane; t < TSEQ; t += 32) {
    const int idx = seq * TSEQ + t;
    if (Y[idx] != IGNORE_IDX) { sum += logp[idx]; cnt += 1.f; }
  }
#pragma unroll
  for (int o = 16; o > 0; o >>= 1) {
    sum += __shfl_down(sum, o);
    cnt += __shfl_down(cnt, o);
  }
  if (lane == 0) { ssum[seq] = sum; scnt[seq] = cnt; }
  __syncthreads();
  if (tid == 0) {
    float avg[SEQS];
#pragma unroll
    for (int s = 0; s < SEQS; ++s) avg[s] = ssum[s] / scnt[s];
    float num = 0.f, den = 0.f;
#pragma unroll
    for (int s = 0; s < SEQS / 2; ++s) { num += ssum[s]; den += scnt[s]; }
    const float nll = -num / den;
    float lmean = 0.f;
#pragma unroll
    for (int s = 0; s < SEQS / 2; ++s) {
      const float d = avg[s] - avg[s + SEQS / 2] - GAMMA_C / BETA_C;
      lmean += log_sigmoid_f(BETA_C * d);
    }
    lmean *= (2.0f / (float)SEQS);
    out[0] = ALPHA_C * nll - lmean;
  }
}

// ---------------------------------------------------------------------------
// Launcher
// ---------------------------------------------------------------------------
extern "C" void kernel_launch(void* const* d_in, const int* in_sizes, int n_in,
                              void* d_out, int out_size, void* d_ws, size_t ws_size,
                              hipStream_t stream) {
  const float* X = (const float*)d_in[0];   // (8,256,4096) fp32
  const float* W = (const float*)d_in[1];   // (32000,4096) fp32
  const int*   Y = (const int*)d_in[2];     // (8,256) int32
  float* out = (float*)d_out;

  // Workspace layout
  size_t off = 0;
  float* part_max = (float*)((char*)d_ws + off); off += (size_t)MTOT * NVT * 4;
  float* part_sum = (float*)((char*)d_ws + off); off += (size_t)MTOT * NVT * 4;
  float* tgt      = (float*)((char*)d_ws + off); off += (size_t)MTOT * 4;
  float* logp     = (float*)((char*)d_ws + off); off += (size_t)MTOT * 4;
  off = (off + 255) & ~(size_t)255;
  const size_t xb_bytes = (size_t)MTOT * HDIM * 2;
  const size_t wb_bytes = (size_t)VDIM * HDIM * 2;
  const bool conv = (off + xb_bytes + wb_bytes) <= ws_size;
  __bf16* Xb = (__bf16*)((char*)d_ws + off);
  __bf16* Wb = (__bf16*)((char*)d_ws + off + xb_bytes);

  if (conv) {
    const long nx = (long)MTOT * HDIM;      // 8,388,608
    const long nw = (long)VDIM * HDIM;      // 131,072,000
    simpo_cvt_bf16<<<(unsigned)(nx / (256 * 4)), 256, 0, stream>>>(X, Xb, nx);
    simpo_cvt_bf16<<<(unsigned)(nw / (256 * 4)), 256, 0, stream>>>(W, Wb, nw);
    simpo_gemm_partials<true><<<dim3(NMT, NVT), 256, 0, stream>>>(
        X, W, Xb, Wb, Y, part_max, part_sum, tgt);
  } else {
    simpo_gemm_partials<false><<<dim3(NMT, NVT), 256, 0, stream>>>(
        X, W, Xb, Wb, Y, part_max, part_sum, tgt);
  }
  simpo_logz<<<MTOT, 256, 0, stream>>>(part_max, part_sum, tgt, logp);
  simpo_final<<<1, 256, 0, stream>>>(logp, Y, out);
}
